// VQActivation_49039936586124
// MI455X (gfx1250) — matrix operand order repaired and checked by reading, not compiled, not run on hardware
//
#include <hip/hip_runtime.h>

typedef __attribute__((ext_vector_type(16))) _Float16 v16h;
typedef __attribute__((ext_vector_type(8)))  float    v8f;

#define RVQ_N     16384
#define RVQ_DIM   512
#define RVQ_KS    4096
#define RVQ_DEPTH 4

#define WAVES           4
#define THREADS         (WAVES * 32)            // 128
#define ROWS_PER_WAVE   32                      // two 16-row A tiles (M blocking)
#define ROWS_PER_BLOCK  (WAVES * ROWS_PER_WAVE) // 128 rows -> 128 blocks, 1/WGP
#define KT              (RVQ_DIM / 32)          // 16 k-tiles of 32
#define CT              (RVQ_KS / 16)           // 256 code tiles of 16
#define TILE_BYTES      (16 * RVQ_DIM * 2)      // 16 KB staged f16 code tile
#define RESID_BYTES     (ROWS_PER_BLOCK * RVQ_DIM * 4)   // 256 KB
#define BPIPE           4                       // B-fragment pipeline depth

// ---------- codebook f32 -> f16 (one-time, 4 MB into d_ws) ----------
__global__ void cvt_f32_to_f16_kernel(const float* __restrict__ in,
                                      unsigned short* __restrict__ out) {
  int i = (blockIdx.x * blockDim.x + threadIdx.x) * 4;
  float4 v = *(const float4*)(in + i);
  union { _Float16 h[4]; uint2 u; } cv;
  cv.h[0] = (_Float16)v.x; cv.h[1] = (_Float16)v.y;
  cv.h[2] = (_Float16)v.z; cv.h[3] = (_Float16)v.w;
  *(uint2*)(out + i) = cv.u;
}

union BFrag { uint4 u[2]; v16h h; };

// CDNA5 async global->LDS DMA (tracked by ASYNCcnt, no VGPR data path).
// Generic pointers into the LDS aperture keep the LDS byte offset in addr[31:0].
__device__ __forceinline__ void stage_fill_16kb(char* dst_lds, const void* src,
                                                int tid) {
  unsigned l = (unsigned)(size_t)(void*)dst_lds;
  const char* g = (const char*)src;
  #pragma unroll
  for (int c = 0; c < 8; ++c) {
    unsigned off = (unsigned)tid * 16u + (unsigned)c * (THREADS * 16u);
    asm volatile("global_load_async_to_lds_b128 %0, %1, off"
                 :: "v"(l + off), "v"((unsigned long long)(g + off))
                 : "memory");
  }
}
__device__ __forceinline__ void wait_async0() {
  asm volatile("s_wait_asynccnt 0x0" ::: "memory");
}

// ---------- fused residual-VQ: all DEPTH steps, rows wave-private ----------
__launch_bounds__(THREADS)
__attribute__((amdgpu_waves_per_eu(1)))
__global__ void rvq_kernel(const float* __restrict__ x,
                           const float* __restrict__ cb32,
                           const unsigned short* __restrict__ cb16,
                           float* __restrict__ out) {
  extern __shared__ char smem_raw[];
  float* resid   = (float*)smem_raw;                       // 256 KB residuals
  char*  stage   = smem_raw + RESID_BYTES;                 // 2 x 16 KB code tiles
  int*   codesAll = (int*)(stage + 2 * TILE_BYTES);

  const int tid  = threadIdx.x;
  const int wave = tid >> 5;
  const int lane = tid & 31;
  const int m16  = lane & 15;
  const int hs   = lane >> 4;

  float* residW = resid + (size_t)wave * (ROWS_PER_WAVE * RVQ_DIM);
  int*   codes  = codesAll + wave * ROWS_PER_WAVE;
  const long rowBase = (long)blockIdx.x * ROWS_PER_BLOCK + wave * ROWS_PER_WAVE;

  // ---- load x tile into LDS residual (f32) ----
  {
    const float4* xs = (const float4*)(x + rowBase * RVQ_DIM);
    float4* rd = (float4*)residW;
    #pragma unroll 4
    for (int i = lane; i < ROWS_PER_WAVE * RVQ_DIM / 4; i += 32) rd[i] = xs[i];
  }

  #pragma unroll 1
  for (int step = 0; step < RVQ_DEPTH; ++step) {
    // kick the DMA for code tile 0 before doing local work (overlap)
    stage_fill_16kb(stage, cb16, tid);

    // ---- build two 16x512 A tiles (f16) from f32 residual, ISA A layout ----
    v16h A0[KT], A1[KT];
    {
      const float* rr0 = residW + m16 * RVQ_DIM + hs * 8;
      const float* rr1 = residW + (m16 + 16) * RVQ_DIM + hs * 8;
      #pragma unroll
      for (int t = 0; t < KT; ++t) {
        v16h a0, a1;
        const float* p0 = rr0 + t * 32;
        const float* p1 = rr1 + t * 32;
        #pragma unroll
        for (int e = 0; e < 8; ++e) {
          a0[e]     = (_Float16)p0[e];       // K = 32t + 8*hs + e
          a0[e + 8] = (_Float16)p0[e + 16];  // K = 32t + 16 + 8*hs + e
          a1[e]     = (_Float16)p1[e];
          a1[e + 8] = (_Float16)p1[e + 16];
        }
        A0[t] = a0; A1[t] = a1;
      }
    }

    float vmax0[8], vmax1[8];
    int   varg0[8], varg1[8];
    #pragma unroll
    for (int r = 0; r < 8; ++r) {
      vmax0[r] = -3.4e38f; varg0[r] = 0;
      vmax1[r] = -3.4e38f; varg1[r] = 0;
    }

    // ---- scan all 4096 codes; shared LDS tile, double-buffered async DMA ----
    #pragma unroll 1
    for (int ct = 0; ct < CT; ++ct) {
      const int par = ct & 1;
      wait_async0();        // own share of buffer[par] landed
      __syncthreads();      // everyone's share landed; buffer[par^1] drained
      if (ct + 1 < CT)
        stage_fill_16kb(stage + (par ^ 1) * TILE_BYTES,
                        cb16 + (size_t)(ct + 1) * 16 * RVQ_DIM, tid);

      const uint4* pb = (const uint4*)(stage + par * TILE_BYTES
                                       + (size_t)m16 * (RVQ_DIM * 2) + hs * 16);
      v8f c0 = {}, c1 = {};

      // B-fragment software pipeline: ds_loads issued BPIPE k-tiles ahead.
      BFrag b[KT];
      #pragma unroll
      for (int i = 0; i < BPIPE; ++i) {
        b[i].u[0] = pb[i * 4];
        b[i].u[1] = pb[i * 4 + 2];
      }
      #pragma unroll
      for (int t = 0; t < KT; ++t) {
        if (t + BPIPE < KT) {
          b[t + BPIPE].u[0] = pb[(t + BPIPE) * 4];
          b[t + BPIPE].u[1] = pb[(t + BPIPE) * 4 + 2];
        }
        c0 = __builtin_amdgcn_wmma_f32_16x16x32_f16(false, A0[t], false, b[t].h,
                                                    (short)0, c0, false, false);
        c1 = __builtin_amdgcn_wmma_f32_16x16x32_f16(false, A1[t], false, b[t].h,
                                                    (short)0, c1, false, false);
      }

      // Pin the pipeline so the scheduler can't re-serialize it:
      // 0x100 = DS-read group, 0x008 = MFMA/WMMA group.
      // DS reads run BPIPE fragments (2 ds_load_b128 each) ahead of the WMMAs,
      // letting the waitcnt pass emit partial s_wait_dscnt (in-order DS returns).
      #pragma unroll
      for (int i = 0; i < BPIPE; ++i)
        __builtin_amdgcn_sched_group_barrier(0x100, 2, 0);
      #pragma unroll
      for (int t = 0; t < KT; ++t) {
        __builtin_amdgcn_sched_group_barrier(0x008, 2, 0);
        if (t + BPIPE < KT)
          __builtin_amdgcn_sched_group_barrier(0x100, 2, 0);
      }

      const int n = ct * 16 + m16;
      #pragma unroll
      for (int r = 0; r < 8; ++r) {
        float v0 = c0[r];
        if (v0 > vmax0[r]) { vmax0[r] = v0; varg0[r] = n; }
        float v1 = c1[r];
        if (v1 > vmax1[r]) { vmax1[r] = v1; varg1[r] = n; }
      }
    }

    // ---- argmax across the 16 lanes of each half-wave (first-max tie-break) ----
    #pragma unroll
    for (int r = 0; r < 8; ++r) {
      #pragma unroll
      for (int off = 8; off >= 1; off >>= 1) {
        float ov0 = __shfl_xor(vmax0[r], off, 32);
        int   oa0 = __shfl_xor(varg0[r], off, 32);
        if (ov0 > vmax0[r] || (ov0 == vmax0[r] && oa0 < varg0[r])) {
          vmax0[r] = ov0; varg0[r] = oa0;
        }
        float ov1 = __shfl_xor(vmax1[r], off, 32);
        int   oa1 = __shfl_xor(varg1[r], off, 32);
        if (ov1 > vmax1[r] || (ov1 == vmax1[r] && oa1 < varg1[r])) {
          vmax1[r] = ov1; varg1[r] = oa1;
        }
      }
    }
    if (m16 == 0) {              // lane 0 -> rows 0..7 / 16..23, lane 16 -> +8
      #pragma unroll
      for (int r = 0; r < 8; ++r) {
        codes[hs * 8 + r]      = varg0[r];
        codes[16 + hs * 8 + r] = varg1[r];
      }
    }

    // ---- exact f32 inner product u + residual update, per row ----
    #pragma unroll 1
    for (int m = 0; m < ROWS_PER_WAVE; ++m) {
      const int code = codes[m];
      const float* crow = cb32 + (size_t)code * RVQ_DIM;
      float* rrow = residW + m * RVQ_DIM;
      float p = 0.f;
      #pragma unroll
      for (int j = 0; j < RVQ_DIM / 32; ++j) {
        int k = lane + j * 32;
        p += rrow[k] * crow[k];
      }
      #pragma unroll
      for (int off = 16; off >= 1; off >>= 1) p += __shfl_xor(p, off, 32);
      #pragma unroll
      for (int j = 0; j < RVQ_DIM / 32; ++j) {
        int k = lane + j * 32;
        rrow[k] -= p * crow[k];
      }
    }
  }

  // ---- total = x - residual_final ----
  {
    const float4* xs = (const float4*)(x + rowBase * RVQ_DIM);
    const float4* rd = (const float4*)residW;
    float4* o = (float4*)(out + rowBase * RVQ_DIM);
    #pragma unroll 4
    for (int i = lane; i < ROWS_PER_WAVE * RVQ_DIM / 4; i += 32) {
      float4 xv = xs[i]; float4 rv = rd[i];
      o[i] = make_float4(xv.x - rv.x, xv.y - rv.y, xv.z - rv.z, xv.w - rv.w);
    }
  }
}

extern "C" void kernel_launch(void* const* d_in, const int* in_sizes, int n_in,
                              void* d_out, int out_size, void* d_ws, size_t ws_size,
                              hipStream_t stream) {
  (void)in_sizes; (void)n_in; (void)out_size; (void)ws_size;
  const float* x    = (const float*)d_in[0];
  const float* cb32 = (const float*)d_in[1];
  unsigned short* cb16 = (unsigned short*)d_ws;   // 4 MB f16 codebook

  cvt_f32_to_f16_kernel<<<RVQ_KS * RVQ_DIM / (256 * 4), 256, 0, stream>>>(cb32, cb16);

  const size_t smem = (size_t)RESID_BYTES + 2 * TILE_BYTES
                    + (size_t)WAVES * ROWS_PER_WAVE * sizeof(int);  // ~288.5 KB
  hipFuncSetAttribute((const void*)rvq_kernel,
                      hipFuncAttributeMaxDynamicSharedMemorySize, (int)smem);
  rvq_kernel<<<RVQ_N / ROWS_PER_BLOCK, THREADS, smem, stream>>>(
      x, cb32, cb16, (float*)d_out);
}
